// CustomLinear_76201309766375
// MI455X (gfx1250) — 8-bit matrix operand order repaired and checked by reading, compiled, not run on hardware
//
#include <hip/hip_runtime.h>
#include <hip/hip_bf16.h>
#include <stdint.h>

typedef __attribute__((ext_vector_type(8))) int   v8i;
typedef __attribute__((ext_vector_type(4))) int   vi4;
typedef __attribute__((ext_vector_type(2))) int   vi2;
typedef __attribute__((ext_vector_type(4))) float vf4;

typedef int ivec4 __attribute__((vector_size(16)));
typedef __attribute__((address_space(1))) ivec4 as1_ivec4;
typedef __attribute__((address_space(3))) ivec4 as3_ivec4;

#if defined(__has_builtin)
#if __has_builtin(__builtin_amdgcn_global_load_async_to_lds_b128)
#define USE_ASYNC_LDS 1
#endif
#endif

__device__ inline void async_copy16(const int8_t* g, int8_t* l) {
#ifdef USE_ASYNC_LDS
    __builtin_amdgcn_global_load_async_to_lds_b128(
        reinterpret_cast<as1_ivec4*>(reinterpret_cast<uintptr_t>(g)),
        reinterpret_cast<as3_ivec4*>(reinterpret_cast<uintptr_t>(l)),
        0, 0);
#else
    *(vi4*)l = *(const vi4*)g;
#endif
}

__device__ inline void async_join() {
#ifdef USE_ASYNC_LDS
#if __has_builtin(__builtin_amdgcn_s_wait_asynccnt)
    __builtin_amdgcn_s_wait_asynccnt(0);
#else
    asm volatile("s_wait_asynccnt 0x0" ::: "memory");
#endif
#endif
}

__device__ inline float wave32_max(float v) {
    v = fmaxf(v, __int_as_float(__builtin_amdgcn_ds_swizzle(__float_as_int(v), 0x401f)));
    v = fmaxf(v, __int_as_float(__builtin_amdgcn_ds_swizzle(__float_as_int(v), 0x201f)));
    v = fmaxf(v, __int_as_float(__builtin_amdgcn_ds_swizzle(__float_as_int(v), 0x101f)));
    v = fmaxf(v, __int_as_float(__builtin_amdgcn_ds_swizzle(__float_as_int(v), 0x081f)));
    v = fmaxf(v, __int_as_float(__builtin_amdgcn_ds_swizzle(__float_as_int(v), 0x041f)));
    return v;
}

__device__ inline uint32_t quant_pack4(vf4 v, float inv) {
    float r0 = fminf(fmaxf(__builtin_rintf(v.x * inv), -128.f), 127.f);
    float r1 = fminf(fmaxf(__builtin_rintf(v.y * inv), -128.f), 127.f);
    float r2 = fminf(fmaxf(__builtin_rintf(v.z * inv), -128.f), 127.f);
    float r3 = fminf(fmaxf(__builtin_rintf(v.w * inv), -128.f), 127.f);
    uint32_t b0 = (uint32_t)(uint8_t)(int)r0;
    uint32_t b1 = (uint32_t)(uint8_t)(int)r1;
    uint32_t b2 = (uint32_t)(uint8_t)(int)r2;
    uint32_t b3 = (uint32_t)(uint8_t)(int)r3;
    return b0 | (b1 << 8) | (b2 << 16) | (b3 << 24);
}

__global__ __launch_bounds__(256) void quant_rows_kernel(
        const float* __restrict__ src, int8_t* __restrict__ q,
        float* __restrict__ scale, int D, float pre_clamp, float post_clamp) {
    const int lane = threadIdx.x & 31;
    const int wave = threadIdx.x >> 5;
    const int row  = blockIdx.x * (blockDim.x >> 5) + wave;
    const float* rsrc = src + (size_t)row * D;
    int8_t*      rq   = q   + (size_t)row * D;
    const int chunks = D >> 7;

    float m = 0.f;
    for (int c = 0; c < chunks; ++c) {
        vf4 v = *(const vf4*)(rsrc + c * 128 + lane * 4);
        m = fmaxf(m, fmaxf(fmaxf(fabsf(v.x), fabsf(v.y)),
                           fmaxf(fabsf(v.z), fabsf(v.w))));
    }
    m = wave32_max(m);
    const float s = fmaxf(fmaxf(m, pre_clamp) * (1.0f / 127.0f), post_clamp);
    if (lane == 0) scale[row] = s;
    const float inv = 1.0f / s;
    for (int c = 0; c < chunks; ++c) {
        vf4 v = *(const vf4*)(rsrc + c * 128 + lane * 4);
        *(uint32_t*)(rq + c * 128 + lane * 4) = quant_pack4(v, inv);
    }
}

__device__ inline v8i wmma_iu8(v8i a, v8i b, v8i c) {
    return __builtin_amdgcn_wmma_i32_16x16x64_iu8(true, a, true, b, c, false, false);
}

__device__ inline v8i load_a_frag(const int8_t* p) {
    vi2 x0 = *(const vi2*)(p +  0);
    vi2 x1 = *(const vi2*)(p + 16);
    vi2 x2 = *(const vi2*)(p + 32);
    vi2 x3 = *(const vi2*)(p + 48);
    v8i a;
    a[0] = x0.x; a[1] = x0.y; a[2] = x1.x; a[3] = x1.y;
    a[4] = x2.x; a[5] = x2.y; a[6] = x3.x; a[7] = x3.y;
    return a;
}

__device__ inline v8i load_b_frag(const int8_t* p) {
    vi2 x0 = *(const vi2*)(p +  0);
    vi2 x1 = *(const vi2*)(p + 16);
    vi2 x2 = *(const vi2*)(p + 32);
    vi2 x3 = *(const vi2*)(p + 48);
    v8i b;
    b[0] = x0.x; b[1] = x0.y; b[2] = x1.x; b[3] = x1.y;
    b[4] = x2.x; b[5] = x2.y; b[6] = x3.x; b[7] = x3.y;
    return b;
}

__global__ __launch_bounds__(256) void int8_wmma_gemm_kernel(
        const int8_t* __restrict__ xq, const int8_t* __restrict__ wq,
        const float* __restrict__ xs, const float* __restrict__ wsc,
        const float* __restrict__ bias, float* __restrict__ out,
        int D, int O) {
    __shared__ __align__(16) int8_t As[2][128 * 64];
    __shared__ __align__(16) int8_t Bs[2][128 * 64];

    const int t      = threadIdx.x;
    const int lane   = t & 31;
    const int wave   = t >> 5;
    const int waveM  = wave & 3;
    const int waveN  = wave >> 2;
    const int half   = lane >> 4;
    const int l15    = lane & 15;
    const int mBlock = blockIdx.x * 128;
    const int nBlock = blockIdx.y * 128;

    const int sRow   = t >> 2;
    const int sCol   = (t & 3) * 16;
    const int sOffLo = sRow * 64 + sCol;
    const int sOffHi = (64 + sRow) * 64 + sCol;
    const int8_t* gA0 = xq + (size_t)(mBlock + sRow) * D + sCol;
    const int8_t* gA1 = xq + (size_t)(mBlock + 64 + sRow) * D + sCol;
    const int8_t* gB0 = wq + (size_t)(nBlock + sRow) * D + sCol;
    const int8_t* gB1 = wq + (size_t)(nBlock + 64 + sRow) * D + sCol;

    const int aFragOff = (waveM * 32 + l15) * 64 + half * 8;
    const int bFragOff = (waveN * 64 + l15) * 64 + half * 8;

    v8i acc[2][4];
#pragma unroll
    for (int mt = 0; mt < 2; ++mt)
#pragma unroll
        for (int nt = 0; nt < 4; ++nt)
            acc[mt][nt] = (v8i){0, 0, 0, 0, 0, 0, 0, 0};

    async_copy16(gA0, &As[0][sOffLo]);
    async_copy16(gA1, &As[0][sOffHi]);
    async_copy16(gB0, &Bs[0][sOffLo]);
    async_copy16(gB1, &Bs[0][sOffHi]);
    gA0 += 64; gA1 += 64; gB0 += 64; gB1 += 64;
    async_join();
    __syncthreads();

    int buf = 0;
    for (int k0 = 0; k0 < D - 64; k0 += 64) {
        const int nb = buf ^ 1;
        async_copy16(gA0, &As[nb][sOffLo]);
        async_copy16(gA1, &As[nb][sOffHi]);
        async_copy16(gB0, &Bs[nb][sOffLo]);
        async_copy16(gB1, &Bs[nb][sOffHi]);
        gA0 += 64; gA1 += 64; gB0 += 64; gB1 += 64;

        v8i a[2], b[4];
#pragma unroll
        for (int mt = 0; mt < 2; ++mt)
            a[mt] = load_a_frag(&As[buf][aFragOff + mt * (16 * 64)]);
#pragma unroll
        for (int nt = 0; nt < 4; ++nt)
            b[nt] = load_b_frag(&Bs[buf][bFragOff + nt * (16 * 64)]);
#pragma unroll
        for (int mt = 0; mt < 2; ++mt)
#pragma unroll
            for (int nt = 0; nt < 4; ++nt)
                acc[mt][nt] = wmma_iu8(a[mt], b[nt], acc[mt][nt]);

        async_join();
        __syncthreads();
        buf ^= 1;
    }

    {
        v8i a[2], b[4];
#pragma unroll
        for (int mt = 0; mt < 2; ++mt)
            a[mt] = load_a_frag(&As[buf][aFragOff + mt * (16 * 64)]);
#pragma unroll
        for (int nt = 0; nt < 4; ++nt)
            b[nt] = load_b_frag(&Bs[buf][bFragOff + nt * (16 * 64)]);
#pragma unroll
        for (int mt = 0; mt < 2; ++mt)
#pragma unroll
            for (int nt = 0; nt < 4; ++nt)
                acc[mt][nt] = wmma_iu8(a[mt], b[nt], acc[mt][nt]);
    }

    const int m0 = mBlock + waveM * 32;
    const int n0 = nBlock + waveN * 64;
#pragma unroll
    for (int nt = 0; nt < 4; ++nt) {
        const int   n   = n0 + nt * 16 + l15;
        const float wsn = wsc[n];
        const float bn  = bias[n];
#pragma unroll
        for (int mt = 0; mt < 2; ++mt) {
            const v8i acv = acc[mt][nt];
#pragma unroll
            for (int r = 0; r < 8; ++r) {
                const int m = m0 + mt * 16 + r + 8 * half;
                out[(size_t)m * O + n] = (float)acv[r] * xs[m] * wsn + bn;
            }
        }
    }
}

extern "C" void kernel_launch(void* const* d_in, const int* in_sizes, int n_in,
                              void* d_out, int out_size, void* d_ws, size_t ws_size,
                              hipStream_t stream) {
    const float* x    = (const float*)d_in[0];
    const float* w    = (const float*)d_in[1];
    const float* bias = (const float*)d_in[2];
    float* out = (float*)d_out;

    const int O = in_sizes[2];
    const int D = in_sizes[1] / O;
    const int M = in_sizes[0] / D;

    char*   wsb = (char*)d_ws;
    int8_t* xq  = (int8_t*)wsb;
    int8_t* wq  = (int8_t*)(wsb + (size_t)M * D);
    float*  xs  = (float*)(wsb + (size_t)M * D + (size_t)O * D);
    float*  wsc = xs + M;

    quant_rows_kernel<<<M / 8, 256, 0, stream>>>(x, xq, xs, D, 0.0f, 1e-12f);
    quant_rows_kernel<<<O / 8, 256, 0, stream>>>(w, wq, wsc, D, 1e-8f, 0.0f);

    dim3 grid(M / 128, O / 128);
    int8_wmma_gemm_kernel<<<grid, 256, 0, stream>>>(xq, wq, xs, wsc, bias, out, D, O);
}
